// DeepJ_45749991637684
// MI455X (gfx1250) — matrix operand order repaired and checked by reading, not compiled, not run on hardware
//
#include <hip/hip_runtime.h>
#include <math.h>

// ---------------- problem constants ----------------
#define HH    512
#define G3H   1536
#define NACT  356
#define SU    32
#define BB    64
#define TT    256
#define MM    (BB*TT)      // 16384 rows (b*T + t)
#define K0P   416          // 388 padded to mult of 32
#define NOUTP 384          // 356 padded to 24 N-tiles

typedef __bf16 bf16;
typedef __attribute__((ext_vector_type(16))) __bf16 bf16x16;
typedef __attribute__((ext_vector_type(8)))  __bf16 bf16x8;
typedef __attribute__((ext_vector_type(8)))  float  f32x8;
typedef __attribute__((ext_vector_type(4)))  unsigned int uint32x4;
typedef __attribute__((ext_vector_type(8)))  int int32x8;
typedef __attribute__((ext_vector_type(4)))  int int32x4;

// ---------------- helpers ----------------
__device__ __forceinline__ float bf2f(bf16 x) { return (float)x; }
__device__ __forceinline__ bf16  f2bf(float x) { return (bf16)x; }

__device__ __forceinline__ f32x8 zero8() {
    f32x8 z;
#pragma unroll
    for (int i = 0; i < 8; ++i) z[i] = 0.0f;
    return z;
}

// A fragment: 16x32 bf16, row-major source. base already points at (m0,k0).
__device__ __forceinline__ bf16x16 load_frag_a(const bf16* base, int lda, int lane) {
    const bf16* p = base + (size_t)(lane & 15) * lda + ((lane >> 4) << 3);
    bf16x8 lo = *(const bf16x8*)p;
    bf16x8 hi = *(const bf16x8*)(p + 16);
    bf16x16 a;
#pragma unroll
    for (int i = 0; i < 8; ++i) { a[i] = lo[i]; a[i + 8] = hi[i]; }
    return a;
}

// B fragment: 32x16 bf16 from weight stored [N,K] row-major. base points at (n0,k0).
__device__ __forceinline__ bf16x16 load_frag_b(const bf16* base, int ldb, int lane) {
    const bf16* p = base + (size_t)(lane & 15) * ldb + ((lane >> 4) << 3);
    bf16x8 lo = *(const bf16x8*)p;
    bf16x8 hi = *(const bf16x8*)(p + 16);
    bf16x16 b;
#pragma unroll
    for (int i = 0; i < 8; ++i) { b[i] = lo[i]; b[i + 8] = hi[i]; }
    return b;
}

__device__ __forceinline__ f32x8 wmma_bf16(bf16x16 a, bf16x16 b, f32x8 c) {
    return __builtin_amdgcn_wmma_f32_16x16x32_bf16(false, a, false, b, (short)0, c, false, false);
}

__device__ __forceinline__ float sigmoidf_(float x) { return 1.0f / (1.0f + __expf(-x)); }

// ---- Tensor Data Mover: 2D bf16 tile (rows x cols elements) global -> LDS ----
// D# layout per CDNA5 ISA (group0 128b, group1 256b).
// This toolchain exposes the 6-arg builtin: (g0, g1, g2, g3, g4, cpol).
__device__ __forceinline__ void tdm_load_2d_bf16(unsigned lds_off, const void* gptr,
                                                 unsigned rows, unsigned cols,
                                                 unsigned row_stride_elems) {
    unsigned long long ga = (unsigned long long)(uintptr_t)gptr;
    uint32x4 g0;
    g0[0] = 1u;                                            // count=1, user mode
    g0[1] = lds_off;                                       // lds_addr (bytes)
    g0[2] = (unsigned)(ga & 0xFFFFFFFFu);                  // global_addr[31:0]
    g0[3] = (unsigned)((ga >> 32) & 0x01FFFFFFu)           // global_addr[56:32]
          | 0x80000000u;                                   // type=2 ("image")
    int32x8 g1;
    g1[0] = (int)(1u << 16);                               // data_size=1 (2 bytes), wg_mask=0
    g1[1] = (int)((cols & 0xFFFFu) << 16);                 // tensor_dim0[15:0]
    g1[2] = (int)(((cols >> 16) & 0xFFFFu)                 // tensor_dim0[31:16]
          | ((rows & 0xFFFFu) << 16));                     // tensor_dim1[15:0]
    g1[3] = (int)(((rows >> 16) & 0xFFFFu)                 // tensor_dim1[31:16]
          | ((cols & 0xFFFFu) << 16));                     // tile_dim0 = cols
    g1[4] = (int)(rows & 0xFFFFu);                         // tile_dim1 = rows, tile_dim2=0
    g1[5] = (int)row_stride_elems;                         // tensor_dim0_stride[31:0]
    g1[6] = 0;                                             // stride hi / dim1_stride lo
    g1[7] = 0;
    int32x4 z4;
    z4[0] = 0; z4[1] = 0; z4[2] = 0; z4[3] = 0;
    int32x8 z8;
#pragma unroll
    for (int i = 0; i < 8; ++i) z8[i] = 0;
    __builtin_amdgcn_tensor_load_to_lds(g0, g1, z4, z4, z8, 0);
}

__device__ __forceinline__ unsigned lds_offset_of(const void* p) {
    return (unsigned)(uintptr_t)p;     // low 32 bits of shared aperture addr = LDS byte offset
}

// ---------------- tiny prep kernels ----------------
__global__ void k_cvt(const float* __restrict__ src, int rs_, int cs,
                      bf16* __restrict__ dst, int rd, int cd) {
    int idx = blockIdx.x * blockDim.x + threadIdx.x;
    if (idx >= rd * cd) return;
    int r = idx / cd, c = idx - r * cd;
    float v = (r < rs_ && c < cs) ? src[(size_t)r * cs + c] : 0.0f;
    dst[idx] = f2bf(v);
}

__global__ void k_style(const float* __restrict__ style, const float* __restrict__ sw,
                        const float* __restrict__ sb, float* __restrict__ s) {
    int idx = blockIdx.x * blockDim.x + threadIdx.x;
    if (idx >= BB * SU) return;
    int b = idx >> 5, j = idx & 31;
    float acc = sb[j];
#pragma unroll
    for (int k = 0; k < 4; ++k) acc += style[b * 4 + k] * sw[j * 4 + k];
    s[idx] = acc;
}

__global__ void k_concat(const float* __restrict__ x, const float* __restrict__ s,
                         bf16* __restrict__ xc) {
    int m = blockIdx.x;             // b*T + t
    int b = m / TT;
    for (int c = threadIdx.x; c < K0P; c += blockDim.x) {
        float v = 0.0f;
        if (c < NACT)            v = x[(size_t)m * NACT + c];
        else if (c < NACT + SU)  v = s[b * SU + (c - NACT)];
        xc[(size_t)m * K0P + c] = f2bf(v);
    }
}

// ---------------- fused input projection: gi = LN1(X W^T + b), l3 = LN3(i_n) ----------------
// grid = MM/16 blocks, 512 threads (16 waves). TDM stages the 16xK A-block into LDS once;
// the same LDS buffer is reused (after a barrier) as the LN staging buffer.
__global__ __launch_bounds__(512)
void k_ih_fused(const bf16* __restrict__ X, int K,
                const bf16* __restrict__ W,                 // [1536, K]
                const float* __restrict__ b_ih,
                const float* __restrict__ g1, const float* __restrict__ be1,
                const float* __restrict__ g3, const float* __restrict__ be3,
                bf16* __restrict__ o_ir, bf16* __restrict__ o_ii, bf16* __restrict__ o_l3) {
    __shared__ __align__(16) bf16 smem[16 * G3H];           // 48KB: A tile, then ghln
    __shared__ float rs[16], rq[16], rs3[16], rq3[16];
    __shared__ float mu1[16], ri1[16], mu3[16], ri3[16];

    const int tid = threadIdx.x;
    const int lane = tid & 31;
    const int wv = tid >> 5;                                // 0..15
    const int m0 = blockIdx.x * 16;
    const int nt0 = wv * 6;
    const int rrow = (lane >> 4) << 3;                      // 0 or 8

    bf16* Atile = smem;                                     // [16][K]
    bf16* ghln  = smem;                                     // [16][G3H] (aliased, used later)

    if (wv == 0) {
        tdm_load_2d_bf16(lds_offset_of(Atile), X + (size_t)m0 * K, 16u, (unsigned)K, (unsigned)K);
        __builtin_amdgcn_s_wait_tensorcnt(0);
    }
    if (tid < 16) { rs[tid] = 0.f; rq[tid] = 0.f; rs3[tid] = 0.f; rq3[tid] = 0.f; }
    __syncthreads();                                        // A tile resident

    f32x8 acc[6];
#pragma unroll
    for (int j = 0; j < 6; ++j) acc[j] = zero8();

    for (int k0 = 0; k0 < K; k0 += 32) {
        bf16x16 a = load_frag_a(Atile + k0, K, lane);       // ds_load_b128
        bf16x16 bfrag[6];
#pragma unroll
        for (int j = 0; j < 6; ++j)                         // issue all 12 b128 loads first
            bfrag[j] = load_frag_b(W + (size_t)(nt0 + j) * 16 * K + k0, K, lane);
#pragma unroll
        for (int j = 0; j < 6; ++j)                         // then drain into WMMAs
            acc[j] = wmma_bf16(a, bfrag[j], acc[j]);
    }

    // bias + LN1 stats (VGPR r belongs to row rrow+r, col = nt*16 + lane%16)
    float ps[8], pq[8];
#pragma unroll
    for (int r = 0; r < 8; ++r) { ps[r] = 0.f; pq[r] = 0.f; }
#pragma unroll
    for (int j = 0; j < 6; ++j) {
        int n = (nt0 + j) * 16 + (lane & 15);
        float bi = b_ih[n];
#pragma unroll
        for (int r = 0; r < 8; ++r) {
            float v = acc[j][r] + bi;
            acc[j][r] = v;
            ps[r] += v; pq[r] += v * v;
        }
    }
#pragma unroll
    for (int r = 0; r < 8; ++r) {
        float s = ps[r], q = pq[r];
#pragma unroll
        for (int m = 1; m < 16; m <<= 1) { s += __shfl_xor(s, m, 32); q += __shfl_xor(q, m, 32); }
        if ((lane & 15) == 0) { atomicAdd(&rs[rrow + r], s); atomicAdd(&rq[rrow + r], q); }
    }
    __syncthreads();                                        // also: last A read before ghln reuse
    if (tid < 16) {
        float mean = rs[tid] * (1.0f / 1536.0f);
        float var  = (rq[tid] - 1536.0f * mean * mean) * (1.0f / 1535.0f);  // unbiased (ddof=1)
        var = fmaxf(var, 0.0f);
        mu1[tid] = mean;
        ri1[tid] = 1.0f / (sqrtf(var) + 1e-6f);
    }
    __syncthreads();

    // normalize into LDS (bf16) + LN3 stats over i_n slice (n>=1024)
    float ps3[8], pq3[8];
#pragma unroll
    for (int r = 0; r < 8; ++r) { ps3[r] = 0.f; pq3[r] = 0.f; }
#pragma unroll
    for (int j = 0; j < 6; ++j) {
        int nt = nt0 + j;
        int n = nt * 16 + (lane & 15);
        float g = g1[n], bv = be1[n];
        bool in3 = (nt >= 64);
#pragma unroll
        for (int r = 0; r < 8; ++r) {
            int row = rrow + r;
            float v = g * (acc[j][r] - mu1[row]) * ri1[row] + bv;
            bf16 vb = f2bf(v);
            ghln[row * G3H + n] = vb;
            if (in3) { float vf = bf2f(vb); ps3[r] += vf; pq3[r] += vf * vf; }
        }
    }
#pragma unroll
    for (int r = 0; r < 8; ++r) {
        float s = ps3[r], q = pq3[r];
#pragma unroll
        for (int m = 1; m < 16; m <<= 1) { s += __shfl_xor(s, m, 32); q += __shfl_xor(q, m, 32); }
        if ((lane & 15) == 0) { atomicAdd(&rs3[rrow + r], s); atomicAdd(&rq3[rrow + r], q); }
    }
    __syncthreads();
    if (tid < 16) {
        float mean = rs3[tid] * (1.0f / 512.0f);
        float var  = (rq3[tid] - 512.0f * mean * mean) * (1.0f / 511.0f);
        var = fmaxf(var, 0.0f);
        mu3[tid] = mean;
        ri3[tid] = 1.0f / (sqrtf(var) + 1e-6f);
    }
    __syncthreads();

    // emit i_r, i_i (LN1 slices) and l3_in = LN3(i_n)
    for (int idx = tid; idx < 16 * HH; idx += 512) {
        int row = idx >> 9, k = idx & 511;
        size_t gm = (size_t)(m0 + row);
        float inn = bf2f(ghln[row * G3H + 2 * HH + k]);
        float l3 = g3[k] * (inn - mu3[row]) * ri3[row] + be3[k];
        o_ir[gm * HH + k] = ghln[row * G3H + k];
        o_ii[gm * HH + k] = ghln[row * G3H + HH + k];
        o_l3[gm * HH + k] = f2bf(l3);
    }
}

// ---------------- persistent GRU recurrence ----------------
// grid = 8 blocks (8 batch rows each), 512 threads (16 waves). Whole T loop inside.
__global__ __launch_bounds__(512)
void k_scan(const bf16* __restrict__ Whh,                   // [1536, 512]
            const float* __restrict__ b_hh,
            const float* __restrict__ g2, const float* __restrict__ be2,
            const float* __restrict__ g4, const float* __restrict__ be4,
            const bf16* __restrict__ i_r, const bf16* __restrict__ i_i,
            const bf16* __restrict__ l3in,
            bf16* __restrict__ h_out,                        // [MM, 512]
            float* __restrict__ hT) {                        // [BB, 512] slice
    __shared__ __align__(16) bf16 hb[16][HH];               // A tile, rows 8..15 stay zero
    __shared__ float hf[8][HH];
    __shared__ bf16 ghb[8][G3H];
    __shared__ float rs[8], rq[8], rs4[8], rq4[8];
    __shared__ float mu2[8], ri2[8], mu4[8], ri4[8];

    const int tid = threadIdx.x;
    const int lane = tid & 31;
    const int wv = tid >> 5;
    const int b0 = blockIdx.x * 8;
    const int nt0 = wv * 6;

    for (int i = tid; i < 16 * HH; i += 512) (&hb[0][0])[i] = f2bf(0.0f);
    for (int i = tid; i < 8 * HH; i += 512) (&hf[0][0])[i] = 0.0f;

    for (int t = 0; t < TT; ++t) {
        if (tid < 8) { rs[tid] = 0.f; rq[tid] = 0.f; rs4[tid] = 0.f; rq4[tid] = 0.f; }
        __syncthreads();   // hb from previous step visible; stats zeroed

        // gh = h @ Whh^T : A from LDS, B streamed from L2-resident Whh
        f32x8 acc[6];
#pragma unroll
        for (int j = 0; j < 6; ++j) acc[j] = zero8();
        for (int k0 = 0; k0 < HH; k0 += 32) {
            bf16x16 a = load_frag_a(&hb[0][0] + k0, HH, lane);
            bf16x16 bfrag[6];
#pragma unroll
            for (int j = 0; j < 6; ++j)
                bfrag[j] = load_frag_b(Whh + (size_t)(nt0 + j) * 16 * HH + k0, HH, lane);
#pragma unroll
            for (int j = 0; j < 6; ++j)
                acc[j] = wmma_bf16(a, bfrag[j], acc[j]);
        }

        // bias + LN2 stats (only rows 0..7, held by lanes 0..15)
        float ps[8], pq[8];
#pragma unroll
        for (int r = 0; r < 8; ++r) { ps[r] = 0.f; pq[r] = 0.f; }
#pragma unroll
        for (int j = 0; j < 6; ++j) {
            int n = (nt0 + j) * 16 + (lane & 15);
            float bi = b_hh[n];
#pragma unroll
            for (int r = 0; r < 8; ++r) {
                float v = acc[j][r] + bi;
                acc[j][r] = v;
                ps[r] += v; pq[r] += v * v;
            }
        }
#pragma unroll
        for (int r = 0; r < 8; ++r) {
            float s = ps[r], q = pq[r];
#pragma unroll
            for (int m = 1; m < 16; m <<= 1) { s += __shfl_xor(s, m, 32); q += __shfl_xor(q, m, 32); }
            if (lane == 0) { atomicAdd(&rs[r], s); atomicAdd(&rq[r], q); }
        }
        __syncthreads();
        if (tid < 8) {
            float mean = rs[tid] * (1.0f / 1536.0f);
            float var  = (rq[tid] - 1536.0f * mean * mean) * (1.0f / 1535.0f);
            var = fmaxf(var, 0.0f);
            mu2[tid] = mean;
            ri2[tid] = 1.0f / (sqrtf(var) + 1e-6f);
        }
        __syncthreads();

        // normalize into ghb + LN4 stats over h_n slice
        float ps4[8], pq4[8];
#pragma unroll
        for (int r = 0; r < 8; ++r) { ps4[r] = 0.f; pq4[r] = 0.f; }
#pragma unroll
        for (int j = 0; j < 6; ++j) {
            int nt = nt0 + j;
            int n = nt * 16 + (lane & 15);
            float g = g2[n], bv = be2[n];
            bool in4 = (nt >= 64);
            if (lane < 16) {
#pragma unroll
                for (int r = 0; r < 8; ++r) {
                    float v = g * (acc[j][r] - mu2[r]) * ri2[r] + bv;
                    bf16 vb = f2bf(v);
                    ghb[r][n] = vb;
                    if (in4) { float vf = bf2f(vb); ps4[r] += vf; pq4[r] += vf * vf; }
                }
            }
        }
#pragma unroll
        for (int r = 0; r < 8; ++r) {
            float s = ps4[r], q = pq4[r];
#pragma unroll
            for (int m = 1; m < 16; m <<= 1) { s += __shfl_xor(s, m, 32); q += __shfl_xor(q, m, 32); }
            if (lane == 0) { atomicAdd(&rs4[r], s); atomicAdd(&rq4[r], q); }
        }
        __syncthreads();
        if (tid < 8) {
            float mean = rs4[tid] * (1.0f / 512.0f);
            float var  = (rq4[tid] - 512.0f * mean * mean) * (1.0f / 511.0f);
            var = fmaxf(var, 0.0f);
            mu4[tid] = mean;
            ri4[tid] = 1.0f / (sqrtf(var) + 1e-6f);
        }
        __syncthreads();

        // gates + state update
        for (int idx = tid; idx < 8 * HH; idx += 512) {
            int row = idx >> 9, k = idx & 511;
            size_t gidx = ((size_t)(b0 + row) * TT + t) * HH + k;
            float hr = bf2f(ghb[row][k]);
            float hi = bf2f(ghb[row][HH + k]);
            float hn = bf2f(ghb[row][2 * HH + k]);
            float ln4 = g4[k] * (hn - mu4[row]) * ri4[row] + be4[k];
            float rg = sigmoidf_(bf2f(i_r[gidx]) + hr);
            float zg = sigmoidf_(bf2f(i_i[gidx]) + hi);
            float ng = tanhf(bf2f(l3in[gidx]) + rg * ln4);
            float hold = hf[row][k];
            float hy = ng + zg * (hold - ng);
            hf[row][k] = hy;
            hb[row][k] = f2bf(hy);
            h_out[gidx] = f2bf(hy);
            if (t == TT - 1) hT[(size_t)(b0 + row) * HH + k] = hy;
        }
        // top-of-loop __syncthreads covers hb WAR for next step's WMMA
    }
}

// ---------------- output projection: y = h W_out^T + b ----------------
// grid = MM/16 blocks, 256 threads (8 waves). TDM stages A [16,512] to LDS; 3 N-tiles/wave.
__global__ __launch_bounds__(256)
void k_out(const bf16* __restrict__ Xf, const bf16* __restrict__ Wo,  // [384,512]
           const float* __restrict__ out_b, float* __restrict__ y) {
    __shared__ __align__(16) bf16 Atile[16][HH];            // 16KB
    const int tid = threadIdx.x;
    const int lane = tid & 31;
    const int wv = tid >> 5;                                // 0..7
    const int m0 = blockIdx.x * 16;

    if (wv == 0) {
        tdm_load_2d_bf16(lds_offset_of(&Atile[0][0]), Xf + (size_t)m0 * HH, 16u, HH, HH);
        __builtin_amdgcn_s_wait_tensorcnt(0);
    }
    __syncthreads();

    f32x8 acc[3];
#pragma unroll
    for (int j = 0; j < 3; ++j) acc[j] = zero8();
    const int nt0 = wv * 3;
    for (int k0 = 0; k0 < HH; k0 += 32) {
        bf16x16 a = load_frag_a(&Atile[0][0] + k0, HH, lane);
        bf16x16 bfrag[3];
#pragma unroll
        for (int j = 0; j < 3; ++j)
            bfrag[j] = load_frag_b(Wo + (size_t)(nt0 + j) * 16 * HH + k0, HH, lane);
#pragma unroll
        for (int j = 0; j < 3; ++j)
            acc[j] = wmma_bf16(a, bfrag[j], acc[j]);
    }
#pragma unroll
    for (int j = 0; j < 3; ++j) {
        int n = (nt0 + j) * 16 + (lane & 15);
        if (n < NACT) {
            float bv = out_b[n];
#pragma unroll
            for (int r = 0; r < 8; ++r) {
                int m = m0 + ((lane >> 4) << 3) + r;
                y[(size_t)m * NACT + n] = acc[j][r] + bv;
            }
        }
    }
}

// ---------------- host ----------------
extern "C" void kernel_launch(void* const* d_in, const int* in_sizes, int n_in,
                              void* d_out, int out_size, void* d_ws, size_t ws_size,
                              hipStream_t stream) {
    (void)n_in; (void)out_size; (void)ws_size;
    auto in = [&](int i) { return (const float*)d_in[i]; };

    const float *x, *style, *out_w, *out_b, *style_w, *style_b;
    const float *w_ih[3], *w_hh[3], *b_ih[3], *b_hh[3];
    const float *g1[3], *be1[3], *g2[3], *be2[3], *g3[3], *be3[3], *g4[3], *be4[3];

    bool insertion = (in_sizes[0] == BB * TT * NACT);   // x first => insertion-order flattening
    if (insertion) {
        x = in(0); style = in(1);
        for (int l = 0; l < 3; ++l) {
            int b = 2 + l * 12;
            w_ih[l] = in(b);     w_hh[l] = in(b + 1);
            b_ih[l] = in(b + 2); b_hh[l] = in(b + 3);
            g1[l] = in(b + 4);  be1[l] = in(b + 5);
            g2[l] = in(b + 6);  be2[l] = in(b + 7);
            g3[l] = in(b + 8);  be3[l] = in(b + 9);
            g4[l] = in(b + 10); be4[l] = in(b + 11);
        }
        out_w = in(38); out_b = in(39); style_w = in(40); style_b = in(41);
    } else {                                            // pytree sorted-key flattening
        for (int l = 0; l < 3; ++l) {
            int b = l * 12;
            b_hh[l] = in(b);     b_ih[l] = in(b + 1);
            be1[l] = in(b + 2);  be2[l] = in(b + 3);
            be3[l] = in(b + 4);  be4[l] = in(b + 5);
            g1[l] = in(b + 6);   g2[l] = in(b + 7);
            g3[l] = in(b + 8);   g4[l] = in(b + 9);
            w_hh[l] = in(b + 10); w_ih[l] = in(b + 11);
        }
        out_b = in(36); out_w = in(37); style_b = in(38); style_w = in(39);
        style = in(40); x = in(41);
    }

    // workspace carve-out
    char* wsp = (char*)d_ws;
    size_t off = 0;
    auto carve = [&](size_t bytes) {
        void* p = wsp + off;
        off = (off + bytes + 255) & ~(size_t)255;
        return p;
    };
    float* s_proj = (float*)carve((size_t)BB * SU * 4);
    bf16* xcat  = (bf16*)carve((size_t)MM * K0P * 2);
    bf16* hbufA = (bf16*)carve((size_t)MM * HH * 2);
    bf16* hbufB = (bf16*)carve((size_t)MM * HH * 2);
    bf16* gir   = (bf16*)carve((size_t)MM * HH * 2);
    bf16* gii   = (bf16*)carve((size_t)MM * HH * 2);
    bf16* gl3   = (bf16*)carve((size_t)MM * HH * 2);
    bf16* wih[3];
    wih[0] = (bf16*)carve((size_t)G3H * K0P * 2);
    wih[1] = (bf16*)carve((size_t)G3H * HH * 2);
    wih[2] = (bf16*)carve((size_t)G3H * HH * 2);
    bf16* whh[3];
    for (int l = 0; l < 3; ++l) whh[l] = (bf16*)carve((size_t)G3H * HH * 2);
    bf16* wout = (bf16*)carve((size_t)NOUTP * HH * 2);

    auto cvt = [&](const float* src, int rsr, int csr, bf16* dst, int rd, int cd) {
        int n = rd * cd;
        k_cvt<<<(n + 255) / 256, 256, 0, stream>>>(src, rsr, csr, dst, rd, cd);
    };
    cvt(w_ih[0], G3H, NACT + SU, wih[0], G3H, K0P);
    cvt(w_ih[1], G3H, HH, wih[1], G3H, HH);
    cvt(w_ih[2], G3H, HH, wih[2], G3H, HH);
    for (int l = 0; l < 3; ++l) cvt(w_hh[l], G3H, HH, whh[l], G3H, HH);
    cvt(out_w, NACT, HH, wout, NOUTP, HH);

    k_style<<<(BB * SU + 255) / 256, 256, 0, stream>>>(style, style_w, style_b, s_proj);
    k_concat<<<MM, 128, 0, stream>>>(x, s_proj, xcat);

    float* yout = (float*)d_out;
    float* hT = yout + (size_t)MM * NACT;

    const bf16* Xin = xcat;
    int K = K0P;
    bf16* houts[3] = {hbufA, hbufB, hbufA};
    for (int l = 0; l < 3; ++l) {
        k_ih_fused<<<MM / 16, 512, 0, stream>>>(Xin, K, wih[l], b_ih[l],
                                                g1[l], be1[l], g3[l], be3[l],
                                                gir, gii, gl3);
        k_scan<<<BB / 8, 512, 0, stream>>>(whh[l], b_hh[l], g2[l], be2[l], g4[l], be4[l],
                                           gir, gii, gl3, houts[l],
                                           hT + (size_t)l * BB * HH);
        Xin = houts[l];
        K = HH;
    }
    k_out<<<dim3(MM / 16), 256, 0, stream>>>(houts[2], wout, out_b, yout);
}
